// SSAaligner_84164179132521
// MI455X (gfx1250) — matrix operand order repaired and checked by reading, not compiled, not run on hardware
//
#include <hip/hip_runtime.h>
#include <stdint.h>

// ---------------------------------------------------------------------------
// MI455X (gfx1250). Heavy stage = 2 batched GEMMs (W 64x512 @ z[b] 512x1024)
// via v_wmma_f32_16x16x32_bf16. HBM floor ~201MB / 23.3TB/s ~= 8.6us, so the
// goal is to keep staging VALU below that: vectorized bf16 transpose staging
// for z, and ASYNC global->LDS (global_load_async_to_lds_b128) for W chunks
// from a pre-converted bf16 copy. x/y (25MB) stay L2-resident (192MB L2) for
// the O(B^2) pairwise-L1 stage.
// ---------------------------------------------------------------------------

typedef __bf16 bf16;
typedef bf16  v16bf __attribute__((ext_vector_type(16)));
typedef bf16  v8bf  __attribute__((ext_vector_type(8)));
typedef bf16  v4bf  __attribute__((ext_vector_type(4)));
typedef bf16  v2bf  __attribute__((ext_vector_type(2)));
typedef float v8f   __attribute__((ext_vector_type(8)));
typedef float v4f   __attribute__((ext_vector_type(4)));

#define NB 48    // batch
#define DK 512   // input dim (K)
#define NE 64    // embed dim (M)
#define NN 1024  // N

#define NT 64    // N tile per block
#define KC 32    // K chunk per WMMA step
#define WROW 40  // padded LDS row stride (bf16 elems) -> 80B, 16B aligned
#define ZROW 40

// ---------------------------------------------------------------------------
// Stage 0: convert W (64x512 f32) -> bf16 once; stays L2-resident.
// ---------------------------------------------------------------------------
__global__ __launch_bounds__(256) void ssa_wconv_kernel(
    const float* __restrict__ W, bf16* __restrict__ Wb)
{
    int idx = (blockIdx.x * 256 + threadIdx.x) * 4;   // 32 blocks cover 32768
    v4f w = *(const v4f*)(W + idx);
    v4bf o;
    o[0] = (bf16)w.x; o[1] = (bf16)w.y; o[2] = (bf16)w.z; o[3] = (bf16)w.w;
    *(v4bf*)(Wb + idx) = o;
}

// ---------------------------------------------------------------------------
// Stage 1: x[b,e,n] = sum_d W[e,d]*z[b,d,n] + bias[e]   (blockIdx.z: 0=x,1=y)
// 256 threads = 8 waves; wave w: N-subtile (w&3)*16, M-half (w>>2)*32.
// ---------------------------------------------------------------------------
__global__ __launch_bounds__(256) void ssa_proj_kernel(
    const float* __restrict__ zx, const float* __restrict__ zy,
    const bf16* __restrict__ Wb, const float* __restrict__ bias,
    float* __restrict__ xo, float* __restrict__ yo)
{
    __shared__ __align__(16) bf16 Wl[NE * WROW];  // 64 e  x 32 k (padded)
    __shared__ __align__(16) bf16 Zl[NT * ZROW];  // 64 n  x 32 k (padded)
    __shared__ float Bl[NE];

    const int tid  = threadIdx.x;
    const int lane = tid & 31;
    const int wv   = tid >> 5;
    const int nsub = wv & 3;           // 0..3 -> n-local base nsub*16
    const int m0   = (wv >> 2) * 32;   // 0 or 32

    const int nbase = blockIdx.x * NT;
    const int b     = blockIdx.y;
    const float* z  = (blockIdx.z == 0) ? zx : zy;
    float* out      = (blockIdx.z == 0) ? xo : yo;
    const float* zb = z + (size_t)b * DK * NN;

    if (tid < NE) Bl[tid] = bias[tid];

    v8f acc0 = {};
    v8f acc1 = {};

    // ISA 16-bit A-layout: lane<16 -> K[0..8)+K[16..24); lane>=16 -> +8.
    const int koff = (lane >> 4) << 3;   // 0 or 8
    // ISA 16-bit B-layout: lane = N, lane<16 -> K[0..8)+K[16..24); lane>=16 -> +8.
    const int bko  = (lane >> 4) << 3;   // 0 or 8
    const int frow = lane & 15;

    // per-thread staging coordinates
    const int we  = tid >> 2;            // W: e row
    const int wkq = (tid & 3) * 8;       // W: 8 bf16 = 16B within chunk
    const uint32_t w_ldsoff =
        (uint32_t)(uintptr_t)&Wl[we * WROW + wkq];  // flat LDS addr low 32 = offset

    const int zn = (tid & 15) * 4;       // z: 4 consecutive n
    const int zk = (tid >> 4) * 2;       // z: k pair (0,2,..,30)

    for (int k0 = 0; k0 < DK; k0 += KC) {
        // ---- stage W chunk (64x32 bf16): async global->LDS, 16B per lane ----
        {
            const bf16* g = Wb + (size_t)we * DK + k0 + wkq;
            asm volatile("global_load_async_to_lds_b128 %0, %1, off"
                         :: "v"(w_ldsoff), "v"(g) : "memory");
        }
        // ---- stage z chunk (32 k x 64 n) transposed: 2(k)x4(n) per thread ----
        {
            const float* zp0 = zb + (size_t)(k0 + zk) * NN + nbase + zn;
            v4f a0 = *(const v4f*)zp0;
            v4f a1 = *(const v4f*)(zp0 + NN);
#pragma unroll
            for (int i = 0; i < 4; ++i) {
                v2bf p;
                p[0] = (bf16)a0[i];
                p[1] = (bf16)a1[i];
                *(v2bf*)&Zl[(zn + i) * ZROW + zk] = p;   // 4B store, k-pair contig
            }
        }
        asm volatile("s_wait_asynccnt 0" ::: "memory");
        __syncthreads();

        // ---- build fragments from LDS (all 16B-aligned vector loads) ----
        union { v16bf v; v8bf h[2]; } afr0, afr1, bfr;

        bfr.h[0] = *(const v8bf*)&Zl[(nsub * 16 + frow) * ZROW + bko];
        bfr.h[1] = *(const v8bf*)&Zl[(nsub * 16 + frow) * ZROW + bko + 16];

        const bf16* wp0 = &Wl[(m0 + frow) * WROW + koff];
        const bf16* wp1 = &Wl[(m0 + 16 + frow) * WROW + koff];
        afr0.h[0] = *(const v8bf*)wp0;
        afr0.h[1] = *(const v8bf*)(wp0 + 16);
        afr1.h[0] = *(const v8bf*)wp1;
        afr1.h[1] = *(const v8bf*)(wp1 + 16);

        acc0 = __builtin_amdgcn_wmma_f32_16x16x32_bf16(
            false, afr0.v, false, bfr.v, (short)0, acc0, false, false);
        acc1 = __builtin_amdgcn_wmma_f32_16x16x32_bf16(
            false, afr1.v, false, bfr.v, (short)0, acc1, false, false);

        __syncthreads();
    }

    // ---- epilogue: C layout -> lane<16: M=r, lane>=16: M=r+8; N=lane&15 ----
    const int nglob = nbase + nsub * 16 + (lane & 15);
    const int ehalf = (lane >> 4) * 8;
    float* ob = out + (size_t)b * NE * NN;
#pragma unroll
    for (int r = 0; r < 8; ++r) {
        int e0 = m0 + r + ehalf;
        ob[(size_t)e0 * NN + nglob] = acc0[r] + Bl[e0];
        int e1 = m0 + 16 + r + ehalf;
        ob[(size_t)e1 * NN + nglob] = acc1[r] + Bl[e1];
    }
}

// ---------------------------------------------------------------------------
// Stage 2: s[i,j,e] = -sum_n |x[i,e,n] - y[j,e,n]|   (x,y L2-resident)
// ---------------------------------------------------------------------------
__global__ __launch_bounds__(256) void ssa_pairdist_kernel(
    const float* __restrict__ x, const float* __restrict__ y,
    float* __restrict__ s)
{
    const int i    = blockIdx.x;
    const int j    = blockIdx.y;
    const int lane = threadIdx.x & 31;
    const int wv   = threadIdx.x >> 5;
    const float* xb = x + (size_t)i * NE * NN;
    const float* yb = y + (size_t)j * NE * NN;

    for (int e = wv; e < NE; e += 8) {
        const float* xr = xb + (size_t)e * NN;
        const float* yr = yb + (size_t)e * NN;
        float p = 0.f;
#pragma unroll
        for (int it = 0; it < NN / 128; ++it) {
            v4f a = *(const v4f*)(xr + it * 128 + lane * 4);
            v4f q = *(const v4f*)(yr + it * 128 + lane * 4);
            p += fabsf(a.x - q.x) + fabsf(a.y - q.y) +
                 fabsf(a.z - q.z) + fabsf(a.w - q.w);
        }
        for (int o = 16; o > 0; o >>= 1) p += __shfl_xor(p, o, 32);
        if (lane == 0) s[(size_t)i * NB * NE + (size_t)j * NE + e] = -p;
    }
}

// ---------------------------------------------------------------------------
// Stage 3: per-e slice: a=softmax_j(s), b2=softmax_i(s), t=a+b2-a*b2,
// accumulate num += t*s, den += t into global accumulators.
// ---------------------------------------------------------------------------
__global__ __launch_bounds__(256) void ssa_softreduce_kernel(
    const float* __restrict__ s, float* __restrict__ accum)
{
    const int e = blockIdx.x;
    __shared__ float sl[NB * NB];
    __shared__ float rmax[NB], rsum[NB], cmax[NB], csum[NB];

    for (int idx = threadIdx.x; idx < NB * NB; idx += 256) {
        int i = idx / NB, j = idx % NB;
        sl[idx] = s[(size_t)i * NB * NE + (size_t)j * NE + e];
    }
    __syncthreads();

    if (threadIdx.x < NB) {                    // rows: softmax over j (axis 1)
        int i = threadIdx.x;
        float m = -3.4e38f;
        for (int j = 0; j < NB; ++j) m = fmaxf(m, sl[i * NB + j]);
        float su = 0.f;
        for (int j = 0; j < NB; ++j) su += __expf(sl[i * NB + j] - m);
        rmax[i] = m; rsum[i] = su;
    } else if (threadIdx.x < 2 * NB) {         // cols: softmax over i (axis 0)
        int j = threadIdx.x - NB;
        float m = -3.4e38f;
        for (int i = 0; i < NB; ++i) m = fmaxf(m, sl[i * NB + j]);
        float su = 0.f;
        for (int i = 0; i < NB; ++i) su += __expf(sl[i * NB + j] - m);
        cmax[j] = m; csum[j] = su;
    }
    __syncthreads();

    float num = 0.f, den = 0.f;
    for (int idx = threadIdx.x; idx < NB * NB; idx += 256) {
        int i = idx / NB, j = idx % NB;
        float sv = sl[idx];
        float a  = __expf(sv - rmax[i]) / rsum[i];
        float b2 = __expf(sv - cmax[j]) / csum[j];
        float t  = a + b2 - a * b2;
        num += t * sv;
        den += t;
    }
    for (int o = 16; o > 0; o >>= 1) {
        num += __shfl_xor(num, o, 32);
        den += __shfl_xor(den, o, 32);
    }
    if ((threadIdx.x & 31) == 0) {
        atomicAdd(&accum[0], num);
        atomicAdd(&accum[1], den);
    }
}

__global__ void ssa_init_kernel(float* accum) {
    if (threadIdx.x < 2) accum[threadIdx.x] = 0.f;
}

__global__ void ssa_finalize_kernel(const float* __restrict__ accum,
                                    float* __restrict__ out) {
    out[0] = accum[0] / accum[1];
}

// ---------------------------------------------------------------------------
extern "C" void kernel_launch(void* const* d_in, const int* in_sizes, int n_in,
                              void* d_out, int out_size, void* d_ws, size_t ws_size,
                              hipStream_t stream)
{
    (void)in_sizes; (void)n_in; (void)out_size; (void)ws_size;

    const float* zx   = (const float*)d_in[0];
    const float* zy   = (const float*)d_in[1];
    const float* W    = (const float*)d_in[2];
    const float* bias = (const float*)d_in[3];

    float* ws    = (float*)d_ws;
    float* x     = ws;                              // 48*64*1024 f32
    float* y     = x + (size_t)NB * NE * NN;        // 48*64*1024 f32
    float* s     = y + (size_t)NB * NE * NN;        // 48*48*64  f32
    bf16*  Wb    = (bf16*)(s + (size_t)NB * NB * NE);   // 64*512 bf16 (16B aligned)
    float* accum = (float*)(Wb + (size_t)NE * DK);      // 2 f32

    hipLaunchKernelGGL(ssa_init_kernel, dim3(1), dim3(32), 0, stream, accum);
    hipLaunchKernelGGL(ssa_wconv_kernel, dim3(NE * DK / 1024), dim3(256), 0, stream,
                       W, Wb);
    hipLaunchKernelGGL(ssa_proj_kernel, dim3(NN / NT, NB, 2), dim3(256), 0, stream,
                       zx, zy, Wb, bias, x, y);
    hipLaunchKernelGGL(ssa_pairdist_kernel, dim3(NB, NB), dim3(256), 0, stream,
                       x, y, s);
    hipLaunchKernelGGL(ssa_softreduce_kernel, dim3(NE), dim3(256), 0, stream,
                       s, accum);
    hipLaunchKernelGGL(ssa_finalize_kernel, dim3(1), dim3(1), 0, stream,
                       accum, (float*)d_out);
}
